// HeteroscedasticMoERouter_87875030876715
// MI455X (gfx1250) — matrix operand order repaired and checked by reading, not compiled, not run on hardware
//
#include <hip/hip_runtime.h>

// Problem constants (match reference)
#define Ntok 4096
#define Dmod 1024
#define Eexp 8
#define Uhid 16
#define Hhid 4096
#define TOPK 2
#define AUX_COEF 0.01f
#define Z_COEF   0.01f

// ---------------- workspace layout (bytes, aligned sections) ----------------
// [u: N floats][scalars][cnt][offs][list: E*N ints][slotw: E*N floats]
// [xb: N*D bf16][hbuf: (N*TOPK+64)*H bf16][wt: E*D*H bf16 (reused for both layers)]
static const size_t OFF_U     = 0;
static const size_t OFF_SC    = OFF_U + (size_t)Ntok * 4;            // 16384
static const size_t OFF_CNT   = OFF_SC + 256;
static const size_t OFF_OFFS  = OFF_CNT + 256;
static const size_t OFF_LIST  = OFF_OFFS + 256;
static const size_t OFF_SLOTW = OFF_LIST + (size_t)Eexp * Ntok * 4;
static const size_t OFF_XB    = OFF_SLOTW + (size_t)Eexp * Ntok * 4;
static const size_t OFF_HBUF  = OFF_XB + (size_t)Ntok * Dmod * 2;
#define HBUF_ROWS (Ntok * TOPK + 64)   // ragged-tail padding rows
static const size_t OFF_WT    = OFF_HBUF + (size_t)HBUF_ROWS * Hhid * 2;
// total ws requirement: OFF_WT + E*D*H*2  (~137 MB)

// ---------------- WMMA types ------------------------------------------------
typedef __attribute__((ext_vector_type(16))) __bf16 v16bf;
typedef __attribute__((ext_vector_type(8)))  float  v8f;

union ABf {
  v16bf v;
  uint4 q[2];
};

// ---------------- scalar helpers -------------------------------------------
// compiler-native fp32 -> bf16 (RNE); lowers to v_cvt_pk_bf16_f32
__device__ __forceinline__ unsigned pk2(float lo, float hi) {
  union { __bf16 h[2]; unsigned u; } r;
  r.h[0] = (__bf16)lo;
  r.h[1] = (__bf16)hi;
  return r.u;
}
__device__ __forceinline__ unsigned short f2bf(float f) {
  union { __bf16 h[2]; unsigned short s[2]; } r;
  r.h[0] = (__bf16)f;
  return r.s[0];
}
__device__ __forceinline__ float gelu_exact(float x) {
  return 0.5f * x * (1.0f + erff(x * 0.70710678118654752f));
}
__device__ __forceinline__ float softplus_f(float x) {
  return (x > 20.0f) ? x : log1pf(expf(x));
}

// ---------------- x -> bf16 -------------------------------------------------
__global__ __launch_bounds__(256) void k_cvt_x(const float* __restrict__ x,
                                               unsigned short* __restrict__ xb) {
  size_t idx = 4ull * (blockIdx.x * 256ull + threadIdx.x);
  if (idx + 3 >= (size_t)Ntok * Dmod) return;
  float4 f = *reinterpret_cast<const float4*>(x + idx);
  uint2 p;
  p.x = pk2(f.x, f.y);
  p.y = pk2(f.z, f.w);
  *reinterpret_cast<uint2*>(xb + idx) = p;
}

// ---------------- weight transpose-convert: W[rows][cols] f32 -> WT[cols][rows] bf16
// grid: (cols/32, rows/32, E), block (32, 8); 32x33 LDS tile keeps both sides coalesced
__global__ __launch_bounds__(256) void k_cvt_w(const float* __restrict__ W,
                                               unsigned short* __restrict__ WT,
                                               int rows, int cols) {
  __shared__ float tile[32][33];
  const int e = blockIdx.z;
  const float* Wp = W + (size_t)e * rows * cols;
  unsigned short* WTp = WT + (size_t)e * rows * cols;
  const int c0 = blockIdx.x * 32, k0 = blockIdx.y * 32;
  const int tx = threadIdx.x, ty = threadIdx.y;
#pragma unroll
  for (int i = 0; i < 4; ++i)
    tile[ty + i * 8][tx] = Wp[(size_t)(k0 + ty + i * 8) * cols + (c0 + tx)];
  __syncthreads();
#pragma unroll
  for (int i = 0; i < 4; ++i)
    WTp[(size_t)(c0 + ty + i * 8) * rows + (k0 + tx)] = f2bf(tile[tx][ty + i * 8]);
}

// ---------------- uncertainty head: u = softplus(gelu(x@w1u+b1u)@w2u+b2u) ---
__global__ __launch_bounds__(256) void k_uncert(const float* __restrict__ x,
                                                const float* __restrict__ w1u,
                                                const float* __restrict__ b1u,
                                                const float* __restrict__ w2u,
                                                const float* __restrict__ b2u,
                                                float* __restrict__ u_out,
                                                float* __restrict__ usum) {
  const int n = blockIdx.x, tid = threadIdx.x;
  float acc[Uhid];
#pragma unroll
  for (int j = 0; j < Uhid; ++j) acc[j] = 0.0f;
  for (int d = tid; d < Dmod; d += 256) {
    float xv = x[(size_t)n * Dmod + d];
    const float* wrow = w1u + (size_t)d * Uhid;
#pragma unroll
    for (int j = 0; j < Uhid; ++j) acc[j] += xv * wrow[j];
  }
  __shared__ float red[256];
  __shared__ float hsum[Uhid];
  for (int j = 0; j < Uhid; ++j) {
    red[tid] = acc[j];
    __syncthreads();
    for (int s = 128; s > 0; s >>= 1) {
      if (tid < s) red[tid] += red[tid + s];
      __syncthreads();
    }
    if (tid == 0) hsum[j] = red[0];
    __syncthreads();
  }
  if (tid == 0) {
    float t = b2u[0];
#pragma unroll
    for (int j = 0; j < Uhid; ++j) t += gelu_exact(hsum[j] + b1u[j]) * w2u[j];
    float uu = softplus_f(t);
    u_out[n] = uu;
    atomicAdd(usum, uu);
  }
}

// ---------------- router: softmax, top-2, expert lists, aux accumulators ----
__global__ __launch_bounds__(256) void k_router(const float* __restrict__ x,
                                                const float* __restrict__ wr,
                                                const float* __restrict__ u,
                                                const float* __restrict__ usum,
                                                float* __restrict__ zsum,
                                                float* __restrict__ usage,
                                                int* __restrict__ cnt,
                                                int* __restrict__ list,
                                                float* __restrict__ slotw) {
  const int n = blockIdx.x, tid = threadIdx.x;
  float acc[Eexp];
#pragma unroll
  for (int e = 0; e < Eexp; ++e) acc[e] = 0.0f;
  for (int d = tid; d < Dmod; d += 256) {
    float xv = x[(size_t)n * Dmod + d];
    const float* wrow = wr + (size_t)d * Eexp;
#pragma unroll
    for (int e = 0; e < Eexp; ++e) acc[e] += xv * wrow[e];
  }
  __shared__ float red[256];
  __shared__ float lg[Eexp];
  for (int e = 0; e < Eexp; ++e) {
    red[tid] = acc[e];
    __syncthreads();
    for (int s = 128; s > 0; s >>= 1) {
      if (tid < s) red[tid] += red[tid + s];
      __syncthreads();
    }
    if (tid == 0) lg[e] = red[0];
    __syncthreads();
  }
  if (tid == 0) {
    float un = u[n] / (usum[0] / (float)Ntok + 1e-8f);
    float l[Eexp], p[Eexp];
    float m = -1e30f;
#pragma unroll
    for (int e = 0; e < Eexp; ++e) {
      l[e] = lg[e] + un * wr[(size_t)Dmod * Eexp + e];
      m = fmaxf(m, l[e]);
    }
    float Z = 0.0f;
#pragma unroll
    for (int e = 0; e < Eexp; ++e) { p[e] = expf(l[e] - m); Z += p[e]; }
    float inv = 1.0f / Z;
#pragma unroll
    for (int e = 0; e < Eexp; ++e) p[e] *= inv;
    float lse = m + logf(Z);
    atomicAdd(zsum, lse * lse);
#pragma unroll
    for (int e = 0; e < Eexp; ++e) atomicAdd(&usage[e], p[e]);
    int i1 = 0;
#pragma unroll
    for (int e = 1; e < Eexp; ++e) if (p[e] > p[i1]) i1 = e;
    int i2 = (i1 == 0) ? 1 : 0;
#pragma unroll
    for (int e = 0; e < Eexp; ++e)
      if (e != i1 && p[e] > p[i2]) i2 = e;
    float s = p[i1] + p[i2];
    float w1 = p[i1] / s, w2 = p[i2] / s;
    int s1 = atomicAdd(&cnt[i1], 1);
    list[i1 * Ntok + s1] = n;
    slotw[i1 * Ntok + s1] = w1;
    int s2 = atomicAdd(&cnt[i2], 1);
    list[i2 * Ntok + s2] = n;
    slotw[i2 * Ntok + s2] = w2;
  }
}

// ---------------- prefix offsets + aux loss scalar --------------------------
__global__ void k_finalize(const float* __restrict__ usum,
                           const float* __restrict__ zsum,
                           const float* __restrict__ usage,
                           const int* __restrict__ cnt,
                           int* __restrict__ offs,
                           float* __restrict__ out_aux) {
  if (threadIdx.x != 0 || blockIdx.x != 0) return;
  (void)usum;
  int off = 0;
  for (int e = 0; e < Eexp; ++e) { offs[e] = off; off += cnt[e]; }
  float lb = 0.0f;
  for (int e = 0; e < Eexp; ++e) {
    float um = usage[e] / (float)Ntok;                     // probs.mean over tokens
    float se = (float)cnt[e] / (float)(Ntok * TOPK);       // selection
    lb += um * se;
  }
  lb *= (float)Eexp;
  out_aux[0] = AUX_COEF * lb + Z_COEF * (zsum[0] / (float)Ntok);
}

// ---------------- FFN pass 1: h = gelu(x @ we1 + be1), gathered -------------
// grid: (H/64, Ntok/64, E), block 128 (4 waves).
// Each wave: M=64 tokens x 16 h-cols; B from transposed bf16 weights (2x b128).
__global__ __launch_bounds__(128) void k_ffn1(const unsigned short* __restrict__ xb,
                                              const unsigned short* __restrict__ wt, // [e][h][d] bf16
                                              const float* __restrict__ be1,
                                              const int* __restrict__ cnt,
                                              const int* __restrict__ offs,
                                              const int* __restrict__ list,
                                              unsigned short* __restrict__ hbuf) {
  const int e = blockIdx.z;
  const int t = blockIdx.y;
  const int count = cnt[e];
  if (t * 64 >= count) return;                 // block-uniform: EXEC stays full
  const int wv = threadIdx.x >> 5;
  const int lane = threadIdx.x & 31;
  const int half = lane >> 4;                  // K-half selector
  const int nidx = lane & 15;                  // A-row / B-col / C-col index
  const int bcol = blockIdx.x * 64 + wv * 16 + nidx;

  const unsigned short* xrow[4];
#pragma unroll
  for (int i = 0; i < 4; ++i) {
    int ls = t * 64 + i * 16 + nidx;
    int token = (ls < count) ? list[e * Ntok + ls] : 0;
    xrow[i] = xb + (size_t)token * Dmod;
  }
  // B: wt[e][bcol][k], contiguous in k
  const unsigned short* wrow =
      wt + (size_t)e * Dmod * Hhid + (size_t)bcol * Dmod + half * 8;

  v8f acc0 = {}, acc1 = {}, acc2 = {}, acc3 = {};
  for (int k0 = 0; k0 < Dmod; k0 += 32) {
    ABf b;
    b.q[0] = *reinterpret_cast<const uint4*>(wrow + k0);
    b.q[1] = *reinterpret_cast<const uint4*>(wrow + k0 + 16);
    ABf a0, a1, a2, a3;
    a0.q[0] = *reinterpret_cast<const uint4*>(xrow[0] + k0 + half * 8);
    a0.q[1] = *reinterpret_cast<const uint4*>(xrow[0] + k0 + 16 + half * 8);
    a1.q[0] = *reinterpret_cast<const uint4*>(xrow[1] + k0 + half * 8);
    a1.q[1] = *reinterpret_cast<const uint4*>(xrow[1] + k0 + 16 + half * 8);
    a2.q[0] = *reinterpret_cast<const uint4*>(xrow[2] + k0 + half * 8);
    a2.q[1] = *reinterpret_cast<const uint4*>(xrow[2] + k0 + 16 + half * 8);
    a3.q[0] = *reinterpret_cast<const uint4*>(xrow[3] + k0 + half * 8);
    a3.q[1] = *reinterpret_cast<const uint4*>(xrow[3] + k0 + 16 + half * 8);
    acc0 = __builtin_amdgcn_wmma_f32_16x16x32_bf16(false, a0.v, false, b.v,
                                                   (short)0, acc0, false, false);
    acc1 = __builtin_amdgcn_wmma_f32_16x16x32_bf16(false, a1.v, false, b.v,
                                                   (short)0, acc1, false, false);
    acc2 = __builtin_amdgcn_wmma_f32_16x16x32_bf16(false, a2.v, false, b.v,
                                                   (short)0, acc2, false, false);
    acc3 = __builtin_amdgcn_wmma_f32_16x16x32_bf16(false, a3.v, false, b.v,
                                                   (short)0, acc3, false, false);
  }

  const int eoffs = offs[e];
  const float bias = be1[(size_t)e * Hhid + bcol];
  v8f accs[4] = {acc0, acc1, acc2, acc3};
#pragma unroll
  for (int i = 0; i < 4; ++i) {
#pragma unroll
    for (int v = 0; v < 8; ++v) {
      int rowL = i * 16 + v + (half ? 8 : 0);  // C/D VGPR layout
      int ls = t * 64 + rowL;
      if (ls < count) {
        float val = gelu_exact(accs[i][v] + bias);
        hbuf[(size_t)(eoffs + ls) * Hhid + bcol] = f2bf(val);
      }
    }
  }
}

// ---------------- FFN pass 2: out += w * (h @ we2 + be2), scattered ---------
// grid: (D/64, Ntok/64, E), block 128
__global__ __launch_bounds__(128) void k_ffn2(const unsigned short* __restrict__ hbuf,
                                              const unsigned short* __restrict__ wt, // [e][d][h] bf16
                                              const float* __restrict__ be2,
                                              const int* __restrict__ cnt,
                                              const int* __restrict__ offs,
                                              const int* __restrict__ list,
                                              const float* __restrict__ slotw,
                                              float* __restrict__ out) {
  const int e = blockIdx.z;
  const int t = blockIdx.y;
  const int count = cnt[e];
  if (t * 64 >= count) return;
  const int wv = threadIdx.x >> 5;
  const int lane = threadIdx.x & 31;
  const int half = lane >> 4;
  const int nidx = lane & 15;
  const int dcol = blockIdx.x * 64 + wv * 16 + nidx;

  const unsigned short* arow[4];
#pragma unroll
  for (int i = 0; i < 4; ++i)
    arow[i] = hbuf + (size_t)(offs[e] + t * 64 + i * 16 + nidx) * Hhid;  // padded
  const unsigned short* wrow =
      wt + (size_t)e * Hhid * Dmod + (size_t)dcol * Hhid + half * 8;

  v8f acc0 = {}, acc1 = {}, acc2 = {}, acc3 = {};
  for (int k0 = 0; k0 < Hhid; k0 += 32) {
    ABf b;
    b.q[0] = *reinterpret_cast<const uint4*>(wrow + k0);
    b.q[1] = *reinterpret_cast<const uint4*>(wrow + k0 + 16);
    ABf a0, a1, a2, a3;
    a0.q[0] = *reinterpret_cast<const uint4*>(arow[0] + k0 + half * 8);
    a0.q[1] = *reinterpret_cast<const uint4*>(arow[0] + k0 + 16 + half * 8);
    a1.q[0] = *reinterpret_cast<const uint4*>(arow[1] + k0 + half * 8);
    a1.q[1] = *reinterpret_cast<const uint4*>(arow[1] + k0 + 16 + half * 8);
    a2.q[0] = *reinterpret_cast<const uint4*>(arow[2] + k0 + half * 8);
    a2.q[1] = *reinterpret_cast<const uint4*>(arow[2] + k0 + 16 + half * 8);
    a3.q[0] = *reinterpret_cast<const uint4*>(arow[3] + k0 + half * 8);
    a3.q[1] = *reinterpret_cast<const uint4*>(arow[3] + k0 + 16 + half * 8);
    acc0 = __builtin_amdgcn_wmma_f32_16x16x32_bf16(false, a0.v, false, b.v,
                                                   (short)0, acc0, false, false);
    acc1 = __builtin_amdgcn_wmma_f32_16x16x32_bf16(false, a1.v, false, b.v,
                                                   (short)0, acc1, false, false);
    acc2 = __builtin_amdgcn_wmma_f32_16x16x32_bf16(false, a2.v, false, b.v,
                                                   (short)0, acc2, false, false);
    acc3 = __builtin_amdgcn_wmma_f32_16x16x32_bf16(false, a3.v, false, b.v,
                                                   (short)0, acc3, false, false);
  }

  const float bias = be2[(size_t)e * Dmod + dcol];
  v8f accs[4] = {acc0, acc1, acc2, acc3};
#pragma unroll
  for (int i = 0; i < 4; ++i) {
#pragma unroll
    for (int v = 0; v < 8; ++v) {
      int rowL = i * 16 + v + (half ? 8 : 0);
      int ls = t * 64 + rowL;
      if (ls < count) {
        int token = list[e * Ntok + ls];
        float w = slotw[e * Ntok + ls];
        atomicAdd(&out[(size_t)token * Dmod + dcol], w * (accs[i][v] + bias));
      }
    }
  }
}

// ---------------- launcher --------------------------------------------------
extern "C" void kernel_launch(void* const* d_in, const int* in_sizes, int n_in,
                              void* d_out, int out_size, void* d_ws, size_t ws_size,
                              hipStream_t stream) {
  (void)in_sizes; (void)n_in; (void)ws_size;
  const float* x   = (const float*)d_in[0];
  const float* w1u = (const float*)d_in[1];
  const float* b1u = (const float*)d_in[2];
  const float* w2u = (const float*)d_in[3];
  const float* b2u = (const float*)d_in[4];
  const float* wr  = (const float*)d_in[5];
  const float* we1 = (const float*)d_in[6];
  const float* be1 = (const float*)d_in[7];
  const float* we2 = (const float*)d_in[8];
  const float* be2 = (const float*)d_in[9];
  float* out = (float*)d_out;

  char* ws = (char*)d_ws;
  float* u_buf = (float*)(ws + OFF_U);
  float* usum  = (float*)(ws + OFF_SC);
  float* zsum  = usum + 1;
  float* usage = usum + 2;
  int*   cnt   = (int*)(ws + OFF_CNT);
  int*   offs  = (int*)(ws + OFF_OFFS);
  int*   list  = (int*)(ws + OFF_LIST);
  float* slotw = (float*)(ws + OFF_SLOTW);
  unsigned short* xb   = (unsigned short*)(ws + OFF_XB);
  unsigned short* hbuf = (unsigned short*)(ws + OFF_HBUF);
  unsigned short* wt   = (unsigned short*)(ws + OFF_WT);  // reused for both layers

  // zero output (scatter-add target) and control region (counters/lists/sums)
  (void)hipMemsetAsync(d_out, 0, (size_t)out_size * sizeof(float), stream);
  (void)hipMemsetAsync(d_ws, 0, OFF_XB, stream);

  k_cvt_x<<<(Ntok * Dmod / 4 + 255) / 256, 256, 0, stream>>>(x, xb);
  k_uncert<<<Ntok, 256, 0, stream>>>(x, w1u, b1u, w2u, b2u, u_buf, usum);
  k_router<<<Ntok, 256, 0, stream>>>(x, wr, u_buf, usum, zsum, usage, cnt, list, slotw);
  k_finalize<<<1, 32, 0, stream>>>(usum, zsum, usage, cnt, offs,
                                   out + (size_t)Ntok * Dmod);

  // layer 1: transpose-convert we1 [e][D][H] -> wt [e][H][D] bf16, then GEMM
  k_cvt_w<<<dim3(Hhid / 32, Dmod / 32, Eexp), dim3(32, 8), 0, stream>>>(
      we1, wt, Dmod, Hhid);
  k_ffn1<<<dim3(Hhid / 64, Ntok / 64, Eexp), 128, 0, stream>>>(
      xb, wt, be1, cnt, offs, list, hbuf);

  // layer 2: reuse the same wt buffer (stream-ordered), we2 [e][H][D] -> wt [e][D][H]
  k_cvt_w<<<dim3(Dmod / 32, Hhid / 32, Eexp), dim3(32, 8), 0, stream>>>(
      we2, wt, Hhid, Dmod);
  k_ffn2<<<dim3(Dmod / 64, Ntok / 64, Eexp), 128, 0, stream>>>(
      hbuf, wt, be2, cnt, offs, list, slotw, out);
}
